// TopKSAE_2551210574470
// MI455X (gfx1250) — matrix operand order repaired and checked by reading, not compiled, not run on hardware
//
#include <hip/hip_runtime.h>
#include <hip/hip_bf16.h>
#include <stdint.h>

#define ACT_DIM   2048
#define DICT_SIZE 32768
#define TOPK      64
#define BATCH     4096

typedef __bf16 bf16_t;
typedef __attribute__((ext_vector_type(16))) __bf16 v16bf;
typedef __attribute__((ext_vector_type(8)))  __bf16 v8bf;
typedef __attribute__((ext_vector_type(8)))  float  v8f;

__device__ __forceinline__ unsigned short f2bf(float f) {
    unsigned u = __float_as_uint(f);
    u += 0x7FFFu + ((u >> 16) & 1u);            // round-to-nearest-even
    return (unsigned short)(u >> 16);
}
__device__ __forceinline__ float bf2f(unsigned short s) {
    return __uint_as_float(((unsigned)s) << 16);
}

union UBF16 { v16bf v; v8bf h[2]; };

// A-fragment: 16x32 bf16. Per lane: 8 bf16 at p, 8 bf16 at p+16
// (lanes 0-15: K=0-7 / 16-23 ; lanes 16-31 (p pre-offset by half*8): K=8-15 / 24-31)
__device__ __forceinline__ v16bf ldA(const bf16_t* p) {
    UBF16 u;
    const v8bf* q = reinterpret_cast<const v8bf*>(p);
    u.h[0] = q[0];
    u.h[1] = q[2];
    return u.v;
}
// B-fragment: 32x16 bf16. Per lane: 8 bf16 at p, 8 bf16 at p+16
// (lanes 0-15: K=0-7 / 16-23 ; lanes 16-31 (p pre-offset by half*8): K=8-15 / 24-31)
__device__ __forceinline__ v16bf ldB(const bf16_t* p) {
    UBF16 u;
    const v8bf* q = reinterpret_cast<const v8bf*>(p);
    u.h[0] = q[0];
    u.h[1] = q[2];
    return u.v;
}

// ---------------------------------------------------------------------------
// Kernel 1: xc = bf16(x - b_dec)   [BATCH, ACT_DIM]
// ---------------------------------------------------------------------------
__global__ void __launch_bounds__(256)
sae_prep_x(const float* __restrict__ x, const float* __restrict__ b_dec,
           unsigned short* __restrict__ xc) {
    size_t i    = (size_t)blockIdx.x * blockDim.x + threadIdx.x;
    size_t base = i * 4;
    float4 v = *(const float4*)(x + base);
    int    c = (int)(base & (ACT_DIM - 1));
    float4 b = *(const float4*)(b_dec + c);
    ushort4 o;
    o.x = f2bf(v.x - b.x);
    o.y = f2bf(v.y - b.y);
    o.z = f2bf(v.z - b.z);
    o.w = f2bf(v.w - b.w);
    *(ushort4*)(xc + base) = o;
}

// ---------------------------------------------------------------------------
// Kernel 2: wencb = bf16(W_enc)   [DICT_SIZE, ACT_DIM] (128 MB -> fits L2)
// ---------------------------------------------------------------------------
__global__ void __launch_bounds__(256)
sae_prep_w(const float* __restrict__ w, unsigned short* __restrict__ wb) {
    size_t i    = (size_t)blockIdx.x * blockDim.x + threadIdx.x;
    size_t base = i * 4;
    float4 v = *(const float4*)(w + base);
    ushort4 o;
    o.x = f2bf(v.x); o.y = f2bf(v.y); o.z = f2bf(v.z); o.w = f2bf(v.w);
    *(ushort4*)(wb + base) = o;
}

// ---------------------------------------------------------------------------
// Kernel 3: wdecT[m,d] = bf16(W_dec[d,m])  (enables contiguous sparse decode)
// ---------------------------------------------------------------------------
__global__ void __launch_bounds__(256)
sae_wdec_transpose(const float* __restrict__ wdec, unsigned short* __restrict__ wdecT) {
    __shared__ float t[32][33];
    const int lx = threadIdx.x & 31;
    const int ly = threadIdx.x >> 5;          // 0..7
    const int m0 = blockIdx.x * 32;           // dict dim
    const int d0 = blockIdx.y * 32;           // act dim
#pragma unroll
    for (int j = 0; j < 4; ++j) {
        int d = ly + j * 8;
        t[d][lx] = wdec[(size_t)(d0 + d) * DICT_SIZE + m0 + lx];
    }
    __syncthreads();
#pragma unroll
    for (int j = 0; j < 4; ++j) {
        int m = ly + j * 8;
        wdecT[(size_t)(m0 + m) * ACT_DIM + d0 + lx] = f2bf(t[lx][m]);
    }
}

// ---------------------------------------------------------------------------
// Kernel 4: encoder GEMM, bf16 WMMA, fused bias + ReLU, software-pipelined.
// pre[b,m] = ReLU( xc[b,:] . wencb[m,:] + b_enc[m] ), written into z region.
// Wave tile: 16(M) x 64(N); block: 8 waves -> 32(M) x 256(N).
// Double-buffered A/B fragments so next step's global_load_b128s are in
// flight while current step's WMMAs execute (no s_wait_loadcnt 0 per WMMA).
// ---------------------------------------------------------------------------
__global__ void __launch_bounds__(256)
sae_encode_wmma(const unsigned short* __restrict__ xcb,
                const unsigned short* __restrict__ wencb,
                const float* __restrict__ b_enc,
                float* __restrict__ pre) {
    const int tid   = threadIdx.x;
    const int wave  = tid >> 5;
    const int lane  = tid & 31;
    const int half  = lane >> 4;        // 0 | 1  (which 16-lane group)
    const int l     = lane & 15;
    const int waveM = wave >> 2;        // 0..1
    const int waveN = wave & 3;         // 0..3
    const int m0 = blockIdx.y * 32 + waveM * 16;
    const int n0 = blockIdx.x * 256 + waveN * 64;

    const bf16_t* xA = reinterpret_cast<const bf16_t*>(xcb);
    const bf16_t* wB = reinterpret_cast<const bf16_t*>(wencb);

    v8f acc[4] = {v8f{}, v8f{}, v8f{}, v8f{}};

    const bf16_t* pa = xA + (size_t)(m0 + l) * ACT_DIM + half * 8;
    const bf16_t* pb[4];
#pragma unroll
    for (int t = 0; t < 4; ++t)
        pb[t] = wB + (size_t)(n0 + t * 16 + l) * ACT_DIM + half * 8;

    // ---- pipeline prologue: buffer 0 <- K-step 0 ----
    v16bf A0 = ldA(pa), A1;
    v16bf B0[4], B1[4];
#pragma unroll
    for (int t = 0; t < 4; ++t) B0[t] = ldB(pb[t]);

    for (int k0 = 0; k0 < ACT_DIM; k0 += 64) {
        // issue loads for K-step k0+32 (always valid: ACT_DIM % 64 == 0)
        A1 = ldA(pa + k0 + 32);
#pragma unroll
        for (int t = 0; t < 4; ++t) B1[t] = ldB(pb[t] + k0 + 32);
        __builtin_prefetch(pa + k0 + 128, 0, 3);   // 2 steps ahead

        // compute on buffer 0 (its loads are oldest in flight)
#pragma unroll
        for (int t = 0; t < 4; ++t)
            acc[t] = __builtin_amdgcn_wmma_f32_16x16x32_bf16(
                false, A0, false, B0[t], (short)0, acc[t], false, false);

        // issue loads for K-step k0+64 (skipped on final iteration)
        if (k0 + 64 < ACT_DIM) {
            A0 = ldA(pa + k0 + 64);
#pragma unroll
            for (int t = 0; t < 4; ++t) B0[t] = ldB(pb[t] + k0 + 64);
        }

        // compute on buffer 1
#pragma unroll
        for (int t = 0; t < 4; ++t)
            acc[t] = __builtin_amdgcn_wmma_f32_16x16x32_bf16(
                false, A1, false, B1[t], (short)0, acc[t], false, false);
    }

    // Epilogue: C/D layout -> lane half: M = r + 8*half, N = l
#pragma unroll
    for (int t = 0; t < 4; ++t) {
        const int n    = n0 + t * 16 + l;
        const float bias = b_enc[n];
#pragma unroll
        for (int r = 0; r < 8; ++r) {
            const int om = m0 + half * 8 + r;
            float v = acc[t][r] + bias;
            pre[(size_t)om * DICT_SIZE + n] = v > 0.0f ? v : 0.0f;
        }
    }
}

// ---------------------------------------------------------------------------
// Kernel 5: exact top-64 per row via 4-pass radix select (values >= 0 after
// ReLU, so uint bit-order == value order). Zeroes non-selected entries of z
// in place and emits a compacted (val, idx) list for the sparse decoder.
// ---------------------------------------------------------------------------
__global__ void __launch_bounds__(256)
sae_topk(float* __restrict__ z, float* __restrict__ tv, int* __restrict__ ti) {
    const int row = blockIdx.x;
    float* zr = z + (size_t)row * DICT_SIZE;
    __shared__ unsigned hist[256];
    __shared__ unsigned s_prefix;
    __shared__ int      s_needed;
    __shared__ int      s_eq;
    __shared__ int      s_cnt;
    const int tid = threadIdx.x;
    if (tid == 0) { s_prefix = 0u; s_needed = TOPK; }

    for (int pass = 3; pass >= 0; --pass) {
        hist[tid] = 0u;
        __syncthreads();
        const unsigned shift = (unsigned)pass * 8u;
        const unsigned hmask = (pass == 3) ? 0u : (0xFFFFFFFFu << (shift + 8u));
        const unsigned pfx   = s_prefix;
        for (int i = tid; i < DICT_SIZE; i += 256) {
            unsigned u = __float_as_uint(zr[i]);
            if ((u & hmask) == pfx) atomicAdd(&hist[(u >> shift) & 255u], 1u);
        }
        __syncthreads();
        if (tid == 0) {
            int need = s_needed;
            for (int b = 255; b >= 0; --b) {
                int c = (int)hist[b];
                if (need <= c) { s_prefix = pfx | ((unsigned)b << shift); break; }
                need -= c;
            }
            s_needed = need;
        }
        __syncthreads();
    }

    const unsigned T = s_prefix;                 // bit pattern of K-th largest
    if (tid == 0) { s_eq = s_needed; s_cnt = 0; }
    __syncthreads();

    for (int i = tid; i < DICT_SIZE; i += 256) {
        float    v = zr[i];
        unsigned u = __float_as_uint(v);
        bool keep = false;
        if (u > T)       keep = true;
        else if (u == T) keep = (atomicSub(&s_eq, 1) > 0);
        if (keep) {
            int p = atomicAdd(&s_cnt, 1);
            if (p < TOPK) { tv[row * TOPK + p] = v; ti[row * TOPK + p] = i; }
        } else if (u != 0u) {
            zr[i] = 0.0f;
        }
    }
}

// ---------------------------------------------------------------------------
// Kernel 6: sparse decode. x_hat[b,:] = b_dec + sum_k val_k * wdecT[idx_k,:]
// List sorted by idx (odd-even sort) -> deterministic FP accumulation order.
// ---------------------------------------------------------------------------
__global__ void __launch_bounds__(256)
sae_decode(const float* __restrict__ tv, const int* __restrict__ ti,
           const unsigned short* __restrict__ wdecT,
           const float* __restrict__ b_dec,
           float* __restrict__ xhat) {
    const int row = blockIdx.x;
    const int tid = threadIdx.x;
    __shared__ float sv[TOPK];
    __shared__ int   si[TOPK];
    if (tid < TOPK) { sv[tid] = tv[row * TOPK + tid]; si[tid] = ti[row * TOPK + tid]; }
    __syncthreads();
    for (int phase = 0; phase < TOPK; ++phase) {
        int p = ((tid & 31) << 1) + (phase & 1);
        if (tid < 32 && p + 1 < TOPK) {
            if (si[p] > si[p + 1]) {
                int   a = si[p]; si[p] = si[p + 1]; si[p + 1] = a;
                float f = sv[p]; sv[p] = sv[p + 1]; sv[p + 1] = f;
            }
        }
        __syncthreads();
    }

    const int d0 = tid * 8;                   // 256 threads * 8 = 2048 dims
    float acc[8];
#pragma unroll
    for (int j = 0; j < 8; ++j) acc[j] = b_dec[d0 + j];

    for (int k = 0; k < TOPK; ++k) {
        float v = sv[k];
        if (v == 0.0f) continue;              // block-uniform branch
        const uint4 w = *(const uint4*)(wdecT + (size_t)si[k] * ACT_DIM + d0);
        const unsigned short* ws = (const unsigned short*)&w;
#pragma unroll
        for (int j = 0; j < 8; ++j) acc[j] = fmaf(v, bf2f(ws[j]), acc[j]);
    }
#pragma unroll
    for (int j = 0; j < 8; ++j) xhat[(size_t)row * ACT_DIM + d0 + j] = acc[j];
}

// ---------------------------------------------------------------------------
extern "C" void kernel_launch(void* const* d_in, const int* in_sizes, int n_in,
                              void* d_out, int out_size, void* d_ws, size_t ws_size,
                              hipStream_t stream) {
    const float* x     = (const float*)d_in[0];
    const float* W_enc = (const float*)d_in[1];
    const float* b_enc = (const float*)d_in[2];
    const float* W_dec = (const float*)d_in[3];
    const float* b_dec = (const float*)d_in[4];

    float* xhat = (float*)d_out;
    float* z    = xhat + (size_t)BATCH * ACT_DIM;   // [BATCH, DICT_SIZE]

    char* ws = (char*)d_ws;
    unsigned short* xcb   = (unsigned short*)ws;                                   // 16 MB
    unsigned short* wencb = (unsigned short*)(ws + (size_t)BATCH * ACT_DIM * 2);   // 128 MB
    unsigned short* wdecT = (unsigned short*)((char*)wencb + (size_t)DICT_SIZE * ACT_DIM * 2); // 128 MB
    float*          tv    = (float*)((char*)wdecT + (size_t)DICT_SIZE * ACT_DIM * 2);          // 1 MB
    int*            ti    = (int*)((char*)tv + (size_t)BATCH * TOPK * 4);                      // 1 MB

    sae_prep_x<<<(BATCH * ACT_DIM) / 1024, 256, 0, stream>>>(x, b_dec, xcb);
    sae_prep_w<<<(DICT_SIZE * ACT_DIM) / 1024, 256, 0, stream>>>(W_enc, wencb);
    sae_wdec_transpose<<<dim3(DICT_SIZE / 32, ACT_DIM / 32), 256, 0, stream>>>(W_dec, wdecT);
    sae_encode_wmma<<<dim3(DICT_SIZE / 256, BATCH / 32), 256, 0, stream>>>(xcb, wencb, b_enc, z);
    sae_topk<<<BATCH, 256, 0, stream>>>(z, tv, ti);
    sae_decode<<<BATCH, 256, 0, stream>>>(tv, ti, wdecT, b_dec, xhat);
}
